// CrossFeatureTransformer_67645734912495
// MI455X (gfx1250) — matrix operand order repaired and checked by reading, not compiled, not run on hardware
//
#include <hip/hip_runtime.h>

// Problem dims
#define BH_   1024   // B*H
#define N_    128
#define C_    256
#define C8_   32
#define HID_  1024
#define EPS_  1e-6f

typedef __attribute__((ext_vector_type(16))) _Float16 v16h;
typedef __attribute__((ext_vector_type(8)))  _Float16 v8h;
typedef __attribute__((ext_vector_type(4)))  _Float16 v4h;
typedef __attribute__((ext_vector_type(8)))  float    v8f;

#define WMMA_F16F32(a,b,c) \
  __builtin_amdgcn_wmma_f32_16x16x32_f16(false,(a),false,(b),(short)0,(c),false,false)

// A-matrix 16x32 f16 fragment (ISA 7.12.2): lane&15 = row, K chunks {hl*8, hl*8+16}
__device__ __forceinline__ v16h fragA16(const _Float16* base, int stride, int lane) {
  const _Float16* p = base + (lane & 15) * stride + ((lane >> 4) << 3);
  v8h lo = *(const v8h*)(p);
  v8h hi = *(const v8h*)(p + 16);
  return __builtin_shufflevector(lo, hi, 0,1,2,3,4,5,6,7,8,9,10,11,12,13,14,15);
}
// B-matrix 32x16 f16 fragment from transposed weight (row n = column of W):
// lane&15 = column, K chunks {hl*8, hl*8+16}
__device__ __forceinline__ v16h fragB16(const _Float16* base, int stride, int lane) {
  const _Float16* p = base + (lane & 15) * stride + ((lane >> 4) << 3);
  v8h lo = *(const v8h*)(p);
  v8h hi = *(const v8h*)(p + 16);
  return __builtin_shufflevector(lo, hi, 0,1,2,3,4,5,6,7,8,9,10,11,12,13,14,15);
}

// ---------------------------------------------------------------------------
// Prep: generic f32 -> f16 transpose.  out[n*Kout + k] = in[k*Nout + n]
__global__ void kPrepT(const float* __restrict__ in, _Float16* __restrict__ out,
                       int Nout, int Kout) {
  int idx = blockIdx.x * 256 + threadIdx.x;
  if (idx < Nout * Kout) {
    int n = idx / Kout, k = idx - n * Kout;
    out[idx] = (_Float16)in[k * Nout + n];
  }
}

// WksT[j][k] = (Wk @ sc_w1)[k][j]   (j<32, k<256)
__global__ void kPrepWks(const float* __restrict__ Wk, const float* __restrict__ scw1,
                         _Float16* __restrict__ WksT) {
  int idx = blockIdx.x * 256 + threadIdx.x;     // 8192
  int j = idx >> 8, k = idx & 255;
  float a = 0.f;
  for (int c = 0; c < 256; ++c) a += Wk[k * 256 + c] * scw1[c * 32 + j];
  WksT[j * 256 + k] = (_Float16)a;
}

// PpT[j][k] = (pos_w2 @ sc_w1)[k][j]   (j<32, k<32)
__global__ void kPrepPp(const float* __restrict__ pw2, const float* __restrict__ scw1,
                        _Float16* __restrict__ PpT) {
  int idx = blockIdx.x * 256 + threadIdx.x;     // 1024
  int j = idx >> 5, k = idx & 31;
  float a = 0.f;
  for (int c = 0; c < 256; ++c) a += pw2[k * 256 + c] * scw1[c * 32 + j];
  PpT[j * 32 + k] = (_Float16)a;
}

// ---------------------------------------------------------------------------
// Kernel A: per (b,h) row: LN1 -> q = xln@Wq -> qs = q@sc_w1  (store qs[1024,32])
__global__ __launch_bounds__(256) void kA_ln_q(
    const float* __restrict__ qin, const float* __restrict__ g1,
    const float* __restrict__ b1, const float* __restrict__ Wq,
    const float* __restrict__ scw1, float* __restrict__ qs) {
  __shared__ float sX[256], sQ[256], red[8];
  const int bh = blockIdx.x, t = threadIdx.x, wv = t >> 5;
  float x = qin[bh * 256 + t];
  float s = x;
  #pragma unroll
  for (int o = 16; o; o >>= 1) s += __shfl_xor(s, o, 32);
  if ((t & 31) == 0) red[wv] = s;
  __syncthreads();
  float tot = 0.f;
  #pragma unroll
  for (int w = 0; w < 8; ++w) tot += red[w];
  const float mean = tot * (1.f / 256.f);
  __syncthreads();
  float d = x - mean, s2 = d * d;
  #pragma unroll
  for (int o = 16; o; o >>= 1) s2 += __shfl_xor(s2, o, 32);
  if ((t & 31) == 0) red[wv] = s2;
  __syncthreads();
  float vtot = 0.f;
  #pragma unroll
  for (int w = 0; w < 8; ++w) vtot += red[w];
  const float inv = rsqrtf(vtot * (1.f / 256.f) + EPS_);
  sX[t] = d * inv * g1[t] + b1[t];
  __syncthreads();
  float acc = 0.f;
  for (int k = 0; k < 256; ++k) acc += sX[k] * Wq[k * 256 + t];   // coalesced in t
  sQ[t] = acc;
  __syncthreads();
  if (t < 32) {
    float a = 0.f;
    for (int k = 0; k < 256; ++k) a += sQ[k] * scw1[k * 32 + t];
    qs[bh * 32 + t] = a;
  }
}

// ---------------------------------------------------------------------------
// Kernel B: one workgroup per (b,h).  8 waves, dynamic LDS ~93KB.
//   s1  = relu(A@WksT + h@PpT + (sc_b1 - qs))            [128,32]
//   aw  = s1@scW2T + sc_b2 ; mask ; softmax over N       per 16-col block
//   vp  = A@WvT + h@posW2T + pos_b2   (= V + pos)
//   agg = sum_n vp * aw
constexpr int SMEM_B = (128*264 + 128*40 + 128*40) * 2 + (1088 + 128) * 4; // 92928

__global__ __launch_bounds__(256) void kB_attn(
    const float* __restrict__ key, const float* __restrict__ rpos,
    const int* __restrict__ vis, const float* __restrict__ qs,
    const _Float16* __restrict__ WksT, const _Float16* __restrict__ PpT,
    const _Float16* __restrict__ WvT, const _Float16* __restrict__ posW2T,
    const _Float16* __restrict__ scW2T, const float* __restrict__ posw1,
    const float* __restrict__ posb1, const float* __restrict__ scb1,
    const float* __restrict__ scb2, const float* __restrict__ posb2,
    float* __restrict__ agg) {
  extern __shared__ char smem[];
  _Float16* sA   = (_Float16*)smem;            // [128][264]
  _Float16* sH   = sA + 128 * 264;             // [128][40]
  _Float16* sS1  = sH + 128 * 40;              // [128][40]
  float*    sQb  = (float*)(sS1 + 128 * 40);   // 32  (sc_b1 - qs)
  float*    sScb2= sQb + 32;                   // 256
  float*    sPb2 = sScb2 + 256;                // 256
  float*    sAgg = sPb2 + 256;                 // 256
  float*    sRed = sAgg + 256;                 // 128 (8 waves x 16 cols)
  float*    sPw1 = sRed + 128;                 // 128 (4x32)
  float*    sPb1 = sPw1 + 128;                 // 32
  int*      sVis = (int*)(sPb1 + 32);          // 128

  const int bh = blockIdx.x, t = threadIdx.x;
  const int wv = t >> 5, lane = t & 31, hl = lane >> 4, ln16 = lane & 15;

  // stream key tile to f16 LDS (float4 loads, packed f16 stores)
  const float4* kp = (const float4*)(key + (size_t)bh * N_ * C_);
  for (int i4 = t; i4 < N_ * C_ / 4; i4 += 256) {
    float4 vd = kp[i4];
    int row = i4 >> 6, col = (i4 & 63) << 2;
    v4h h4 = {(_Float16)vd.x, (_Float16)vd.y, (_Float16)vd.z, (_Float16)vd.w};
    *(v4h*)(sA + row * 264 + col) = h4;
  }
  if (t < 32)  sQb[t] = scb1[t] - qs[bh * 32 + t];
  if (t < 128) sPw1[t] = posw1[t];
  if (t >= 128 && t < 160) sPb1[t - 128] = posb1[t - 128];
  if (t < 128) sVis[t] = vis[bh * 128 + t];
  sScb2[t] = scb2[t];
  sPb2[t]  = posb2[t];
  sAgg[t]  = 0.f;
  __syncthreads();

  // pos-MLP hidden h = relu(rp @ pos_w1 + pos_b1)  [128,32]
  for (int idx = t; idx < 128 * 32; idx += 256) {
    int row = idx >> 5, j = idx & 31;
    const float* rp = rpos + ((size_t)bh * 128 + row) * 4;
    float hv = sPb1[j] + rp[0]*sPw1[j] + rp[1]*sPw1[32+j] + rp[2]*sPw1[64+j] + rp[3]*sPw1[96+j];
    sH[row * 40 + j] = (_Float16)fmaxf(hv, 0.f);
  }
  __syncthreads();

  const int rbase = wv * 16 + hl * 8;   // this lane's first D row
  // s1: wave wv handles row block wv; two 16-col blocks
  {
    const _Float16* aB = sA + (wv * 16) * 264;
    const _Float16* hB = sH + (wv * 16) * 40;
    #pragma unroll
    for (int jb = 0; jb < 2; ++jb) {
      v8f acc = {};
      #pragma unroll
      for (int kk = 0; kk < 8; ++kk)
        acc = WMMA_F16F32(fragA16(aB + kk * 32, 264, lane),
                          fragB16(WksT + (jb * 16) * 256 + kk * 32, 256, lane), acc);
      acc = WMMA_F16F32(fragA16(hB, 40, lane),
                        fragB16(PpT + (jb * 16) * 32, 32, lane), acc);
      int j = jb * 16 + ln16;
      float bias = sQb[j];
      #pragma unroll
      for (int i = 0; i < 8; ++i)
        sS1[(rbase + i) * 40 + j] = (_Float16)fmaxf(acc[i] + bias, 0.f);
    }
  }
  __syncthreads();

  // main loop over 16-column blocks
  for (int nj = 0; nj < 16; ++nj) {
    const int n = nj * 16 + ln16;
    // scores
    v8f sc = {};
    sc = WMMA_F16F32(fragA16(sS1 + (wv * 16) * 40, 40, lane),
                     fragB16(scW2T + (nj * 16) * 32, 32, lane), sc);
    const float bias = sScb2[n];
    float aw[8], mmax = -3.4e38f;
    #pragma unroll
    for (int i = 0; i < 8; ++i) {
      float a = sc[i] + bias;
      if (sVis[rbase + i] == 0) a = -1e9f;
      aw[i] = a; mmax = fmaxf(mmax, a);
    }
    mmax = fmaxf(mmax, __shfl_xor(mmax, 16, 32));
    if (hl == 0) sRed[wv * 16 + ln16] = mmax;
    __syncthreads();
    float cmax = sRed[ln16];
    #pragma unroll
    for (int w = 1; w < 8; ++w) cmax = fmaxf(cmax, sRed[w * 16 + ln16]);
    __syncthreads();
    float ssum = 0.f;
    #pragma unroll
    for (int i = 0; i < 8; ++i) { aw[i] = __expf(aw[i] - cmax); ssum += aw[i]; }
    ssum += __shfl_xor(ssum, 16, 32);
    if (hl == 0) sRed[wv * 16 + ln16] = ssum;
    __syncthreads();
    float csum = 0.f;
    #pragma unroll
    for (int w = 0; w < 8; ++w) csum += sRed[w * 16 + ln16];
    __syncthreads();
    const float inv = 1.f / csum;
    // vp = V + pos (chained WMMA accumulation)
    v8f vp = {};
    const _Float16* aB = sA + (wv * 16) * 264;
    #pragma unroll
    for (int kk = 0; kk < 8; ++kk)
      vp = WMMA_F16F32(fragA16(aB + kk * 32, 264, lane),
                       fragB16(WvT + (nj * 16) * 256 + kk * 32, 256, lane), vp);
    vp = WMMA_F16F32(fragA16(sH + (wv * 16) * 40, 40, lane),
                     fragB16(posW2T + (nj * 16) * 32, 32, lane), vp);
    const float pb = sPb2[n];
    float part = 0.f;
    #pragma unroll
    for (int i = 0; i < 8; ++i) part += (vp[i] + pb) * aw[i] * inv;
    part += __shfl_xor(part, 16, 32);
    if (hl == 0) atomicAdd(&sAgg[n], part);   // ds_add_f32
  }
  __syncthreads();
  agg[bh * 256 + t] = sAgg[t];
}

// ---------------------------------------------------------------------------
// Kernel C: 16-row tiles. x = agg@Wo + bo + query; LN2; FF via WMMA; +x
__global__ __launch_bounds__(256) void kC_tail(
    const float* __restrict__ aggf, const float* __restrict__ qin,
    const _Float16* __restrict__ WoT, const float* __restrict__ bo,
    const float* __restrict__ g2, const float* __restrict__ b2,
    const _Float16* __restrict__ fW1T, const float* __restrict__ fb1,
    const _Float16* __restrict__ fW2T, const float* __restrict__ fb2,
    float* __restrict__ out) {
  __shared__ _Float16 sAg[16 * 264];      // agg f16, reused as LN2 output (f16)
  __shared__ float    sX[16 * 264];       // x (with residual), f32
  __shared__ _Float16 sHid[16 * 1032];    // FF hidden f16
  __shared__ float    sRS[256], sRQ[256], sMean[16], sInv[16];

  const int r0 = blockIdx.x * 16, t = threadIdx.x;
  const int wv = t >> 5, lane = t & 31, hl = lane >> 4, ln16 = lane & 15;

  for (int idx = t; idx < 16 * 256; idx += 256) {
    int r = idx >> 8, c = idx & 255;
    sAg[r * 264 + c] = (_Float16)aggf[(r0 + r) * 256 + c];
  }
  __syncthreads();

  // stage 1: x = agg@Wo + bo + query
  #pragma unroll
  for (int jj = 0; jj < 2; ++jj) {
    const int nj = wv * 2 + jj;
    v8f acc = {};
    #pragma unroll
    for (int kk = 0; kk < 8; ++kk)
      acc = WMMA_F16F32(fragA16(sAg + kk * 32, 264, lane),
                        fragB16(WoT + (nj * 16) * 256 + kk * 32, 256, lane), acc);
    const int n = nj * 16 + ln16;
    const float bb = bo[n];
    #pragma unroll
    for (int i = 0; i < 8; ++i) {
      int r = i + hl * 8;
      sX[r * 264 + n] = acc[i] + bb + qin[(r0 + r) * 256 + n];
    }
  }
  __syncthreads();

  // stage 2: per-row LN2 -> sAg (f16)
  {
    const int r = t >> 4, seg = t & 15;
    float ss = 0.f, sq = 0.f;
    #pragma unroll
    for (int c = seg * 16; c < seg * 16 + 16; ++c) {
      float v = sX[r * 264 + c]; ss += v; sq += v * v;
    }
    sRS[t] = ss; sRQ[t] = sq;
  }
  __syncthreads();
  if (t < 16) {
    float S = 0.f, Q = 0.f;
    #pragma unroll
    for (int s = 0; s < 16; ++s) { S += sRS[t * 16 + s]; Q += sRQ[t * 16 + s]; }
    float m = S * (1.f / 256.f);
    sMean[t] = m; sInv[t] = rsqrtf(Q * (1.f / 256.f) - m * m + EPS_);
  }
  __syncthreads();
  for (int idx = t; idx < 16 * 256; idx += 256) {
    int r = idx >> 8, c = idx & 255;
    float y = (sX[r * 264 + c] - sMean[r]) * sInv[r] * g2[c] + b2[c];
    sAg[r * 264 + c] = (_Float16)y;
  }
  __syncthreads();

  // stage 3: hid = relu(y@ff_w1 + fb1)   [16,1024]
  #pragma unroll
  for (int p = 0; p < 8; ++p) {
    const int jb = wv * 8 + p;
    v8f acc = {};
    #pragma unroll
    for (int kk = 0; kk < 8; ++kk)
      acc = WMMA_F16F32(fragA16(sAg + kk * 32, 264, lane),
                        fragB16(fW1T + (jb * 16) * 256 + kk * 32, 256, lane), acc);
    const int j = jb * 16 + ln16;
    const float bb = fb1[j];
    #pragma unroll
    for (int i = 0; i < 8; ++i)
      sHid[(i + hl * 8) * 1032 + j] = (_Float16)fmaxf(acc[i] + bb, 0.f);
  }
  __syncthreads();

  // stage 4: out = hid@ff_w2 + fb2 + x
  #pragma unroll
  for (int jj = 0; jj < 2; ++jj) {
    const int nj = wv * 2 + jj;
    v8f acc = {};
    #pragma unroll
    for (int kk = 0; kk < 32; ++kk)
      acc = WMMA_F16F32(fragA16(sHid + kk * 32, 1032, lane),
                        fragB16(fW2T + (nj * 16) * 1024 + kk * 32, 1024, lane), acc);
    const int n = nj * 16 + ln16;
    const float bb = fb2[n];
    #pragma unroll
    for (int i = 0; i < 8; ++i) {
      int r = i + hl * 8;
      out[(r0 + r) * 256 + n] = acc[i] + bb + sX[r * 264 + n];
    }
  }
}

// ---------------------------------------------------------------------------
// Workspace layout (bytes)
constexpr size_t OFF_QS   = 0;                          // [1024,32] f32
constexpr size_t OFF_AGG  = OFF_QS  + 1024 * 32 * 4;    // [1024,256] f32
constexpr size_t OFF_WVT  = OFF_AGG + 1024 * 256 * 4;   // [256,256] f16
constexpr size_t OFF_WOT  = OFF_WVT + 256 * 256 * 2;    // [256,256] f16
constexpr size_t OFF_WKS  = OFF_WOT + 256 * 256 * 2;    // [32,256] f16
constexpr size_t OFF_PPT  = OFF_WKS + 32 * 256 * 2;     // [32,32] f16
constexpr size_t OFF_PW2T = OFF_PPT + 32 * 32 * 2;      // [256,32] f16
constexpr size_t OFF_SW2T = OFF_PW2T + 256 * 32 * 2;    // [256,32] f16
constexpr size_t OFF_FW1T = OFF_SW2T + 256 * 32 * 2;    // [1024,256] f16
constexpr size_t OFF_FW2T = OFF_FW1T + 1024 * 256 * 2;  // [256,1024] f16

extern "C" void kernel_launch(void* const* d_in, const int* in_sizes, int n_in,
                              void* d_out, int out_size, void* d_ws, size_t ws_size,
                              hipStream_t stream) {
  const float* qin  = (const float*)d_in[0];
  const float* key  = (const float*)d_in[1];
  const float* rpos = (const float*)d_in[2];
  const int*   vis  = (const int*)d_in[3];
  const float* g1   = (const float*)d_in[4];
  const float* b1   = (const float*)d_in[5];
  const float* g2   = (const float*)d_in[6];
  const float* b2   = (const float*)d_in[7];
  const float* Wq   = (const float*)d_in[8];
  const float* Wk   = (const float*)d_in[9];
  const float* Wv   = (const float*)d_in[10];
  const float* pw1  = (const float*)d_in[11];
  const float* pb1  = (const float*)d_in[12];
  const float* pw2  = (const float*)d_in[13];
  const float* pb2  = (const float*)d_in[14];
  const float* sw1  = (const float*)d_in[15];
  const float* sb1  = (const float*)d_in[16];
  const float* sw2  = (const float*)d_in[17];
  const float* sb2  = (const float*)d_in[18];
  const float* Wo   = (const float*)d_in[19];
  const float* bo   = (const float*)d_in[20];
  const float* fw1  = (const float*)d_in[21];
  const float* fb1  = (const float*)d_in[22];
  const float* fw2  = (const float*)d_in[23];
  const float* fb2  = (const float*)d_in[24];
  float* outp = (float*)d_out;

  char* w = (char*)d_ws;
  float*     qs   = (float*)(w + OFF_QS);
  float*     agg  = (float*)(w + OFF_AGG);
  _Float16*  WvT  = (_Float16*)(w + OFF_WVT);
  _Float16*  WoT  = (_Float16*)(w + OFF_WOT);
  _Float16*  WksT = (_Float16*)(w + OFF_WKS);
  _Float16*  PpT  = (_Float16*)(w + OFF_PPT);
  _Float16*  pW2T = (_Float16*)(w + OFF_PW2T);
  _Float16*  sW2T = (_Float16*)(w + OFF_SW2T);
  _Float16*  fW1T = (_Float16*)(w + OFF_FW1T);
  _Float16*  fW2T = (_Float16*)(w + OFF_FW2T);

  // prep: transposed f16 weights (+ fused Wks / Pp)
  kPrepT<<<(256*256+255)/256, 256, 0, stream>>>(Wv,  WvT,  256, 256);
  kPrepT<<<(256*256+255)/256, 256, 0, stream>>>(Wo,  WoT,  256, 256);
  kPrepT<<<(256*32 +255)/256, 256, 0, stream>>>(pw2, pW2T, 256, 32);
  kPrepT<<<(256*32 +255)/256, 256, 0, stream>>>(sw2, sW2T, 256, 32);
  kPrepT<<<(1024*256+255)/256,256, 0, stream>>>(fw1, fW1T, 1024, 256);
  kPrepT<<<(256*1024+255)/256,256, 0, stream>>>(fw2, fW2T, 256, 1024);
  kPrepWks<<<32, 256, 0, stream>>>(Wk, sw1, WksT);
  kPrepPp<<<4, 256, 0, stream>>>(pw2, sw1, PpT);

  // LN1 + q + qs
  kA_ln_q<<<BH_, 256, 0, stream>>>(qin, g1, b1, Wq, sw1, qs);

  // main attention (93KB dynamic LDS; gfx1250 WGP has 320KB)
  hipFuncSetAttribute((const void*)kB_attn,
                      hipFuncAttributeMaxDynamicSharedMemorySize, SMEM_B);
  kB_attn<<<BH_, 256, SMEM_B, stream>>>(key, rpos, vis, qs, WksT, PpT, WvT,
                                        pW2T, sW2T, pw1, pb1, sb1, sb2, pb2, agg);

  // tail: Wo + residual + LN2 + FF (WMMA)
  kC_tail<<<BH_ / 16, 256, 0, stream>>>(agg, qin, WoT, bo, g2, b2,
                                        fW1T, fb1, fW2T, fb2, outp);
}
